// Task_Specific_Attention_28939489640685
// MI455X (gfx1250) — matrix operand order repaired and checked by reading, not compiled, not run on hardware
//
#include <hip/hip_runtime.h>
#include <hip/hip_bf16.h>

// ---------------------------------------------------------------------------
// Problem constants (match the JAX reference) + padded dims
// ---------------------------------------------------------------------------
static constexpr int NB = 128;   // batch
static constexpr int NX = 512;   // Nx
static constexpr int NT = 512;   // Nt
static constexpr int KK = 256;   // K
static constexpr int PP = 86;    // P
static constexpr int DD = 658;   // D
static constexpr int DP   = 672; // D padded to multiple of 32
static constexpr int PPAD = 128; // P padded to multiple of 64

typedef __attribute__((ext_vector_type(16))) __bf16 v16bf;
typedef __attribute__((ext_vector_type(8)))  float  v8f;
typedef __attribute__((ext_vector_type(4)))  unsigned int u32x4;
typedef __attribute__((ext_vector_type(8)))  int          i32x8;
typedef __attribute__((ext_vector_type(4)))  int          i32x4;

// LDS pitches (elements) = row + 8 pad; *2 bytes stays 16B-aligned per row and
// matches TDM pad_amount = 4 DWORDs (16B) after every row.
static constexpr int AP  = 40;   // A row-major [64 m][32 k]
static constexpr int APT = 72;   // A k-major   [32 k][64 m]
static constexpr int BPT = 40;   // B n-major   [128 n][32 k]
static constexpr int BPN = 136;  // B k-major   [32 k][128 n]

#if defined(__HIP_DEVICE_COMPILE__) && __has_builtin(__builtin_amdgcn_tensor_load_to_lds) && __has_builtin(__builtin_amdgcn_s_wait_tensorcnt)
#define HAVE_TDM 1
#else
#define HAVE_TDM 0
#endif

#if HAVE_TDM
// ---------------------------------------------------------------------------
// Issue one 2D TDM tile load Global->LDS (bf16 elements).
//   d0: contiguous extent (elements), d1: #rows, strideElems: row stride,
//   padCode: pad_interval code with 2^(padCode+1) DWORDs per row
//            (row bytes = d0*2; pad 4 DWORDs -> LDS pitch = d0+8 elements)
// D# packing per CDNA5 ISA ch.8 (group0 128b, group1 256b; groups 2/3 unused).
// ---------------------------------------------------------------------------
__device__ __forceinline__ void tdm_load_tile_2d(unsigned ldsAddr, const __bf16* gsrc,
                                                 unsigned d0, unsigned d1,
                                                 unsigned long long strideElems,
                                                 unsigned padCode) {
    unsigned long long ga = (unsigned long long)(const void*)gsrc;
    u32x4 g0;
    g0[0] = 1u;                                   // count=1 (valid user D#)
    g0[1] = ldsAddr;                              // lds_addr (bytes)
    g0[2] = (unsigned)ga;                         // global_addr[31:0]
    g0[3] = (unsigned)(ga >> 32) | (2u << 30);    // global_addr[56:32] | type=2
    i32x8 g1;
    g1[0] = (int)((1u << 16)                      // data_size = 1 -> 2 bytes
                | (1u << 20)                      // pad_enable
                | (padCode << 22)                 // pad_interval
                | (3u << 25));                    // pad_amount = 4 DWORDs
    g1[1] = (int)((d0 & 0xFFFFu) << 16);                          // tensor_dim0 lo16
    g1[2] = (int)(((d0 >> 16) & 0xFFFFu) | ((d1 & 0xFFFFu) << 16)); // td0 hi | td1 lo
    g1[3] = (int)(((d1 >> 16) & 0xFFFFu) | (d0 << 16));           // td1 hi | tile_dim0
    g1[4] = (int)(d1 & 0xFFFFu);                                  // tile_dim1 | tile_dim2=0
    g1[5] = (int)(unsigned)(strideElems & 0xFFFFFFFFull);         // dim0_stride lo32
    g1[6] = (int)(unsigned)((strideElems >> 32) & 0xFFFFull);     // dim0_stride hi16
    g1[7] = 0;
    i32x4 z4 = {0, 0, 0, 0};
#if __clang_major__ >= 23
    i32x8 z8 = {0, 0, 0, 0, 0, 0, 0, 0};
    __builtin_amdgcn_tensor_load_to_lds(g0, g1, z4, z4, z8, 0);
#else
    __builtin_amdgcn_tensor_load_to_lds(g0, g1, z4, z4, 0);
#endif
}
#endif

// ---------------------------------------------------------------------------
// Fragment loaders, wave32: element s of lane half h holds k = 16h + s (contiguous, same for A and B).
// ---------------------------------------------------------------------------
__device__ __forceinline__ v16bf frag_a_rowmajor(const __bf16* As, int mbase) {
    int lane = threadIdx.x & 31, row = mbase + (lane & 15), half = lane >> 4;
    v16bf a;
#pragma unroll
    for (int j = 0; j < 8; ++j) {
        int kb = half * 16 + j * 2;
        a[2 * j]     = As[row * AP + kb];
        a[2 * j + 1] = As[row * AP + kb + 1];
    }
    return a;
}
__device__ __forceinline__ v16bf frag_a_kmajor(const __bf16* At, int mbase) {
    int lane = threadIdx.x & 31, row = mbase + (lane & 15), half = lane >> 4;
    v16bf a;
#pragma unroll
    for (int j = 0; j < 8; ++j) {
        int kb = half * 16 + j * 2;
        a[2 * j]     = At[kb * APT + row];
        a[2 * j + 1] = At[(kb + 1) * APT + row];
    }
    return a;
}
__device__ __forceinline__ v16bf frag_b_nmajor(const __bf16* Bt, int nbase) {
    int lane = threadIdx.x & 31, col = nbase + (lane & 15), half = lane >> 4;
    v16bf b;
#pragma unroll
    for (int j = 0; j < 8; ++j) {
        int k = half * 16 + j * 2;
        b[2 * j]     = Bt[col * BPT + k];
        b[2 * j + 1] = Bt[col * BPT + k + 1];
    }
    return b;
}
__device__ __forceinline__ v16bf frag_b_kmajor(const __bf16* Bs, int nbase) {
    int lane = threadIdx.x & 31, col = nbase + (lane & 15), half = lane >> 4;
    v16bf b;
#pragma unroll
    for (int j = 0; j < 8; ++j) {
        int k = half * 16 + j * 2;
        b[2 * j]     = Bs[k * BPN + col];
        b[2 * j + 1] = Bs[(k + 1) * BPN + col];
    }
    return b;
}

__device__ __forceinline__ void cp16(__bf16* dst, const __bf16* src) {
    *(uint4*)dst       = *(const uint4*)src;
    *(uint4*)(dst + 8) = *(const uint4*)(src + 8);
}

// ---------------------------------------------------------------------------
// Tiled bf16 WMMA GEMM: Out = A(MxK)*B(KxN), f32 accumulate.
// Block = 128 thr (4 waves); block tile 64x128; wave tile 32x64 (2x4 frags);
// K-step 32. ALT: A m-contiguous (k-major LDS); BLT: B k-contiguous (n-major).
// Staging via Tensor Data Mover (double-buffered, TENSORcnt) when available.
// All dims padded -> no bounds checks.
// MODE 0: store bf16. MODE 1: store tanh() bf16.
// MODE 2: atomically accumulate sum_n tanh(acc+Bias)*Wred[n] into Acc[m].
// ---------------------------------------------------------------------------
template <int MODE, bool ALT, bool BLT>
__global__ void __launch_bounds__(128)
gemm_bf16_wmma(const __bf16* __restrict__ A, long sMa, long sKa, long aBatch,
               const __bf16* __restrict__ Bm, long sKb, long sNb, long bBatch,
               __bf16* __restrict__ Out, long ldo, long oBatch,
               const __bf16* __restrict__ Bias, long ldb, long biasBatch,
               const float* __restrict__ Wred,
               float* __restrict__ Acc, long accBatch,
               int Kd) {
    __shared__ __bf16 ldsA[2][64 * AP];    // >= 32*APT
    __shared__ __bf16 ldsB[2][128 * BPT];  // >= 32*BPN

    const int t    = threadIdx.x;
    const int lane = t & 31;
    const int wave = t >> 5;                    // 0..3
    const int bz   = blockIdx.z;
    const int tileM = blockIdx.x * 64;
    const int tileN = blockIdx.y * 128;
    const int wm = (wave >> 1) * 32;
    const int wn = (wave & 1) * 64;

    const __bf16* Ab = A  + (long)bz * aBatch;
    const __bf16* Bb = Bm + (long)bz * bBatch;

    v8f acc[2][4];
#pragma unroll
    for (int mi = 0; mi < 2; ++mi)
#pragma unroll
        for (int ni = 0; ni < 4; ++ni)
            acc[mi][ni] = (v8f){0.f, 0.f, 0.f, 0.f, 0.f, 0.f, 0.f, 0.f};

    const int kIters = Kd >> 5;

#if HAVE_TDM
    unsigned aAddr[2], bAddr[2];
    aAddr[0] = (unsigned)(unsigned long long)(void*)&ldsA[0][0];
    aAddr[1] = (unsigned)(unsigned long long)(void*)&ldsA[1][0];
    bAddr[0] = (unsigned)(unsigned long long)(void*)&ldsB[0][0];
    bAddr[1] = (unsigned)(unsigned long long)(void*)&ldsB[1][0];
    // prologue: DMA tiles for kt=0 into buffer 0 (wave 0 issues; EXEC ignored)
    if (wave == 0) {
        if (!ALT) tdm_load_tile_2d(aAddr[0], Ab + (long)tileM * sMa, 32, 64, (unsigned long long)sMa, 3);
        else      tdm_load_tile_2d(aAddr[0], Ab + tileM,            64, 32, (unsigned long long)sKa, 4);
        if (BLT)  tdm_load_tile_2d(bAddr[0], Bb + (long)tileN * sNb, 32, 128, (unsigned long long)sNb, 3);
        else      tdm_load_tile_2d(bAddr[0], Bb + tileN,            128, 32, (unsigned long long)sKb, 5);
    }
    for (int kt = 0; kt < kIters; ++kt) {
        const int cur = kt & 1;
        if (wave == 0) {
            if (kt + 1 < kIters) {   // DMA next tiles into other buffer
                const int k0n = (kt + 1) * 32;
                const int nxt = cur ^ 1;
                if (!ALT) tdm_load_tile_2d(aAddr[nxt], Ab + (long)tileM * sMa + k0n, 32, 64, (unsigned long long)sMa, 3);
                else      tdm_load_tile_2d(aAddr[nxt], Ab + (long)k0n * sKa + tileM, 64, 32, (unsigned long long)sKa, 4);
                if (BLT)  tdm_load_tile_2d(bAddr[nxt], Bb + (long)tileN * sNb + k0n, 32, 128, (unsigned long long)sNb, 3);
                else      tdm_load_tile_2d(bAddr[nxt], Bb + (long)k0n * sKb + tileN, 128, 32, (unsigned long long)sKb, 5);
                __builtin_amdgcn_s_wait_tensorcnt((short)2);   // 2 newest in flight
            } else {
                __builtin_amdgcn_s_wait_tensorcnt((short)0);
            }
        }
        __syncthreads();   // current tiles resident in LDS for all waves
        const __bf16* As = &ldsA[cur][0];
        const __bf16* Bs = &ldsB[cur][0];
        v16bf a0 = ALT ? frag_a_kmajor(As, wm) : frag_a_rowmajor(As, wm);
        v16bf a1 = ALT ? frag_a_kmajor(As, wm + 16) : frag_a_rowmajor(As, wm + 16);
#pragma unroll
        for (int ni = 0; ni < 4; ++ni) {
            v16bf b = BLT ? frag_b_nmajor(Bs, wn + ni * 16) : frag_b_kmajor(Bs, wn + ni * 16);
            acc[0][ni] = __builtin_amdgcn_wmma_f32_16x16x32_bf16(false, a0, false, b, (short)0, acc[0][ni], false, false);
            acc[1][ni] = __builtin_amdgcn_wmma_f32_16x16x32_bf16(false, a1, false, b, (short)0, acc[1][ni], false, false);
        }
        __syncthreads();   // compute done; buffer may be overwritten next iter
    }
#else
    for (int kt = 0; kt < kIters; ++kt) {
        const int k0 = kt * 32;
        if (!ALT) {   // A k-contiguous: 64 rows x 2 chunks, 1 per thread
            int row = t >> 1, ch = t & 1;
            const __bf16* src = Ab + (long)(tileM + row) * sMa + k0 + ch * 16;
            cp16(&ldsA[0][row * AP + ch * 16], src);
            __builtin_prefetch(src + 32, 0, 0);
        } else {      // A m-contiguous: 32 k x 4 chunks, 1 per thread
            int k = t & 31, mc = t >> 5;
            const __bf16* src = Ab + (long)(k0 + k) * sKa + tileM + mc * 16;
            cp16(&ldsA[0][k * APT + mc * 16], src);
        }
        if (BLT) {    // B k-contiguous: 128 n rows x 2 chunks, 2 per thread
            const __bf16* src = Bb + (long)(tileN + t) * sNb + k0;
            cp16(&ldsB[0][t * BPT], src);
            cp16(&ldsB[0][t * BPT + 16], src + 16);
            __builtin_prefetch(src + 32, 0, 0);
        } else {      // B n-contiguous: 32 k x 8 chunks, 2 per thread
            int k = t & 31, nc = t >> 5;
            const __bf16* src = Bb + (long)(k0 + k) * sKb + tileN;
            cp16(&ldsB[0][k * BPN + nc * 16], src + nc * 16);
            cp16(&ldsB[0][k * BPN + (nc + 4) * 16], src + (nc + 4) * 16);
            __builtin_prefetch(src + (long)32 * sKb, 0, 0);
        }
        __syncthreads();
        const __bf16* As = &ldsA[0][0];
        const __bf16* Bs = &ldsB[0][0];
        v16bf a0 = ALT ? frag_a_kmajor(As, wm) : frag_a_rowmajor(As, wm);
        v16bf a1 = ALT ? frag_a_kmajor(As, wm + 16) : frag_a_rowmajor(As, wm + 16);
#pragma unroll
        for (int ni = 0; ni < 4; ++ni) {
            v16bf b = BLT ? frag_b_nmajor(Bs, wn + ni * 16) : frag_b_kmajor(Bs, wn + ni * 16);
            acc[0][ni] = __builtin_amdgcn_wmma_f32_16x16x32_bf16(false, a0, false, b, (short)0, acc[0][ni], false, false);
            acc[1][ni] = __builtin_amdgcn_wmma_f32_16x16x32_bf16(false, a1, false, b, (short)0, acc[1][ni], false, false);
        }
        __syncthreads();
    }
#endif

    const int col  = lane & 15;
    const int half = lane >> 4;

    if (MODE == 0 || MODE == 1) {
        __bf16* Ob = Out + (long)bz * oBatch;
#pragma unroll
        for (int mi = 0; mi < 2; ++mi)
#pragma unroll
            for (int ni = 0; ni < 4; ++ni)
#pragma unroll
                for (int r = 0; r < 8; ++r) {
                    int m = tileM + wm + mi * 16 + r + half * 8;
                    int n = tileN + wn + ni * 16 + col;
                    float v = acc[mi][ni][r];
                    if (MODE == 1) v = tanhf(v);
                    Ob[(long)m * ldo + n] = (__bf16)v;
                }
    } else {
        const __bf16* Biasb = Bias + (long)bz * biasBatch;
        float* Accb = Acc + (long)bz * accBatch;
        float w[4];
#pragma unroll
        for (int ni = 0; ni < 4; ++ni) w[ni] = Wred[tileN + wn + ni * 16 + col];
#pragma unroll
        for (int mi = 0; mi < 2; ++mi) {
            float rows[8];
#pragma unroll
            for (int r = 0; r < 8; ++r) {
                int m = tileM + wm + mi * 16 + r + half * 8;
                float s = 0.0f;
#pragma unroll
                for (int ni = 0; ni < 4; ++ni) {
                    int n = tileN + wn + ni * 16 + col;
                    s += tanhf(acc[mi][ni][r] + (float)Biasb[(long)m * ldb + n]) * w[ni];
                }
                rows[r] = s;
            }
#pragma unroll
            for (int mask = 1; mask < 16; mask <<= 1)
#pragma unroll
                for (int r = 0; r < 8; ++r)
                    rows[r] += __shfl_xor(rows[r], mask, 32);
            if (col == 0) {
#pragma unroll
                for (int r = 0; r < 8; ++r) {
                    int m = tileM + wm + mi * 16 + r + half * 8;
                    atomicAdd(&Accb[m], rows[r]);
                }
            }
        }
    }
}

// ---------------------------------------------------------------------------
// f32 -> bf16 with 2D zero padding: grid = (rowsOut, ceil(colsOut/256))
// ---------------------------------------------------------------------------
__global__ void pad_cvt_bf16(const float* __restrict__ in, __bf16* __restrict__ out,
                             int rowsIn, int colsIn, int colsOut) {
    int r = blockIdx.x;
    int c = blockIdx.y * blockDim.x + threadIdx.x;
    if (c >= colsOut) return;
    float v = 0.0f;
    if (r < rowsIn && c < colsIn) v = in[(long)r * colsIn + c];
    out[(long)r * colsOut + c] = (__bf16)v;
}

__global__ void zero_f32(float* __restrict__ p, long n) {
    long i = (long)blockIdx.x * blockDim.x + threadIdx.x;
    long stride = (long)gridDim.x * blockDim.x;
    for (; i < n; i += stride) p[i] = 0.0f;
}

// ---------------------------------------------------------------------------
// Per-batch softmax over a_c_weight / a_p_weight + weighted gathers (f32).
// ---------------------------------------------------------------------------
__global__ void softmax_finish(const float* __restrict__ acw, const float* __restrict__ apw,
                               const float* __restrict__ x, const float* __restrict__ tgt,
                               float* __restrict__ cOut, float* __restrict__ pOut) {
    __shared__ float probs_c[NX];
    __shared__ float probs_p[NT];
    __shared__ float red[256];
    const int b = blockIdx.x;
    const int tid = threadIdx.x;

    {
        const float* wrow = acw + (long)b * NX;
        float v0 = wrow[tid], v1 = wrow[tid + 256];
        red[tid] = fmaxf(v0, v1);
        __syncthreads();
        for (int s = 128; s > 0; s >>= 1) { if (tid < s) red[tid] = fmaxf(red[tid], red[tid + s]); __syncthreads(); }
        float mx = red[0];
        __syncthreads();
        float e0 = __expf(v0 - mx), e1 = __expf(v1 - mx);
        red[tid] = e0 + e1;
        __syncthreads();
        for (int s = 128; s > 0; s >>= 1) { if (tid < s) red[tid] += red[tid + s]; __syncthreads(); }
        float inv = 1.0f / red[0];
        __syncthreads();
        probs_c[tid] = e0 * inv;
        probs_c[tid + 256] = e1 * inv;
    }
    {
        const float* wrow = apw + (long)b * NT;
        float v0 = wrow[tid], v1 = wrow[tid + 256];
        red[tid] = fmaxf(v0, v1);
        __syncthreads();
        for (int s = 128; s > 0; s >>= 1) { if (tid < s) red[tid] = fmaxf(red[tid], red[tid + s]); __syncthreads(); }
        float mx = red[0];
        __syncthreads();
        float e0 = __expf(v0 - mx), e1 = __expf(v1 - mx);
        red[tid] = e0 + e1;
        __syncthreads();
        for (int s = 128; s > 0; s >>= 1) { if (tid < s) red[tid] += red[tid + s]; __syncthreads(); }
        float inv = 1.0f / red[0];
        __syncthreads();
        probs_p[tid] = e0 * inv;
        probs_p[tid + 256] = e1 * inv;
    }
    __syncthreads();

    const float* xb = x + (long)b * NX * DD;
    for (int d = tid; d < DD; d += 256) {
        float s = 0.0f;
        for (int i = 0; i < NX; ++i) s += probs_c[i] * xb[(long)i * DD + d];
        cOut[(long)b * DD + d] = s;
    }
    const float* tb = tgt + (long)b * NT * PP;
    for (int d = tid; d < PP; d += 256) {
        float s = 0.0f;
        for (int i = 0; i < NT; ++i) s += probs_p[i] * tb[(long)i * PP + d];
        pOut[(long)b * PP + d] = s;
    }
}

// ---------------------------------------------------------------------------
// Launch
// ---------------------------------------------------------------------------
extern "C" void kernel_launch(void* const* d_in, const int* in_sizes, int n_in,
                              void* d_out, int out_size, void* d_ws, size_t ws_size,
                              hipStream_t stream) {
    (void)in_sizes; (void)n_in; (void)out_size; (void)ws_size;

    const float* x    = (const float*)d_in[0];  // [B,NX,D]
    const float* tgt  = (const float*)d_in[1];  // [B,NT,P]
    const float* W_b  = (const float*)d_in[2];  // [P,D]
    const float* W_x  = (const float*)d_in[3];  // [K,D]
    const float* W_p  = (const float*)d_in[4];  // [K,P]
    const float* w_hx = (const float*)d_in[5];  // [K,1]
    const float* w_hp = (const float*)d_in[6];  // [K,1]

    float* out  = (float*)d_out;
    float* cOut = out;                         // [B,D]
    float* pOut = cOut + (long)NB * DD;        // [B,P]
    float* acw  = pOut + (long)NB * PP;        // [B,1,NX]
    float* apw  = acw  + (long)NB * NX;        // [B,1,NT]

    char* ws = (char*)d_ws;
    size_t off = 0;
    auto alloc = [&](size_t bytes) -> void* {
        void* p = ws + off;
        off = (off + bytes + 255) & ~(size_t)255;
        return p;
    };
    __bf16* Xb  = (__bf16*)alloc((size_t)NB * NX * DP * 2);    // [B,NX,DP]
    __bf16* Tb  = (__bf16*)alloc((size_t)NB * NT * PPAD * 2);  // [B,NT,PPAD]
    __bf16* Wbb = (__bf16*)alloc((size_t)PPAD * DP * 2);       // [PPAD,DP]
    __bf16* Wxb = (__bf16*)alloc((size_t)KK * DP * 2);         // [K,DP]
    __bf16* Wpb = (__bf16*)alloc((size_t)KK * PPAD * 2);       // [K,PPAD]
    __bf16* U   = (__bf16*)alloc((size_t)NB * NX * PPAD * 2);  // X @ W_b^T
    __bf16* Cb  = (__bf16*)alloc((size_t)NB * NT * NX * 2);    // tanh(T @ U^T)
    __bf16* Vb  = (__bf16*)alloc((size_t)NB * NX * KK * 2);    // X @ W_x^T
    __bf16* Sb  = (__bf16*)alloc((size_t)NB * NT * KK * 2);    // T @ W_p^T

    pad_cvt_bf16<<<dim3(NB * NX, 3), 256, 0, stream>>>(x,   Xb,  NB * NX, DD, DP);
    pad_cvt_bf16<<<dim3(NB * NT, 1), 256, 0, stream>>>(tgt, Tb,  NB * NT, PP, PPAD);
    pad_cvt_bf16<<<dim3(PPAD, 3),    256, 0, stream>>>(W_b, Wbb, PP, DD, DP);
    pad_cvt_bf16<<<dim3(KK, 3),      256, 0, stream>>>(W_x, Wxb, KK, DD, DP);
    pad_cvt_bf16<<<dim3(KK, 1),      256, 0, stream>>>(W_p, Wpb, KK, PP, PPAD);
    zero_f32<<<256, 256, 0, stream>>>(acw, (long)NB * (NX + NT));

    // --- G1: U[NX,PPAD] = X[NX,DP] @ Wbb^T ---
    gemm_bf16_wmma<0, false, true><<<dim3(8, 1, NB), 128, 0, stream>>>(
        Xb,  DP, 1, (long)NX * DP,
        Wbb, 1, DP, 0,
        U,   PPAD, (long)NX * PPAD,
        nullptr, 0, 0, nullptr, nullptr, 0, DP);

    // --- G2: C[NT,NX] = tanh(T[NT,PPAD] @ U^T) ---
    gemm_bf16_wmma<1, false, true><<<dim3(8, 4, NB), 128, 0, stream>>>(
        Tb, PPAD, 1, (long)NT * PPAD,
        U,  1, PPAD, (long)NX * PPAD,
        Cb, NX, (long)NT * NX,
        nullptr, 0, 0, nullptr, nullptr, 0, PPAD);

    // --- G3: V[NX,K] = X @ Wxb^T ---
    gemm_bf16_wmma<0, false, true><<<dim3(8, 2, NB), 128, 0, stream>>>(
        Xb,  DP, 1, (long)NX * DP,
        Wxb, 1, DP, 0,
        Vb,  KK, (long)NX * KK,
        nullptr, 0, 0, nullptr, nullptr, 0, DP);

    // --- G4: S[NT,K] = T @ Wpb^T ---
    gemm_bf16_wmma<0, false, true><<<dim3(8, 2, NB), 128, 0, stream>>>(
        Tb,  PPAD, 1, (long)NT * PPAD,
        Wpb, 1, PPAD, 0,
        Sb,  KK, (long)NT * KK,
        nullptr, 0, 0, nullptr, nullptr, 0, PPAD);

    // --- G5: acw[x] += sum_k tanh(V[x,k] + (C^T @ S)[x,k]) * w_hx[k] ---
    gemm_bf16_wmma<2, true, false><<<dim3(8, 2, NB), 128, 0, stream>>>(
        Cb, 1, NX, (long)NT * NX,          // A = C^T (m-contiguous)
        Sb, KK, 1, (long)NT * KK,          // B = S   (n-contiguous)
        nullptr, 0, 0,
        Vb, KK, (long)NX * KK,             // bias = V
        w_hx, acw, NX, NT);

    // --- G6: apw[t] += sum_k tanh(S[t,k] + (C @ V)[t,k]) * w_hp[k] ---
    gemm_bf16_wmma<2, false, false><<<dim3(8, 2, NB), 128, 0, stream>>>(
        Cb, NX, 1, (long)NT * NX,          // A = C   (k-contiguous)
        Vb, KK, 1, (long)NX * KK,          // B = V   (n-contiguous)
        nullptr, 0, 0,
        Sb, KK, (long)NT * KK,             // bias = S
        w_hp, apw, NT, NX);

    softmax_finish<<<NB, 256, 0, stream>>>(acw, apw, x, tgt, cOut, pOut);
}
